// Decoder_32203664786033
// MI455X (gfx1250) — matrix operand order repaired and checked by reading, not compiled, not run on hardware
//
#include <hip/hip_runtime.h>
#include <hip/hip_bf16.h>

// ---------------------------------------------------------------------------
// CDNA5 (gfx1250) LSTM attention decoder.
// Heavy GEMMs: V_WMMA_F32_16X16X32_BF16 (bf16 in, f32 accumulate), with a
// depth-2 software pipeline (two k-tiles of fragments in flight) so the
// waitcnt pass can issue partial s_wait_loadcnt and L2 latency is overlapped
// with matrix-core work.  Weights live in a B-fragment-native packed bf16
// layout (32B contiguous per lane per 32x16 tile), ~49 MB -> L2 resident.
// ---------------------------------------------------------------------------

typedef __attribute__((ext_vector_type(16))) __bf16 v16bf;
typedef __attribute__((ext_vector_type(8)))  __bf16 v8bf;
typedef __attribute__((ext_vector_type(8)))  float  v8f;

#define T_STEPS 96
#define BATCH   64
#define SRC     48
#define HID     1024
#define EMB     1024

// ---- WMMA fragment helpers -------------------------------------------------

// ap points at X + row*ldx + half*8 (row/lane-half baked in); k-tile stride is
// 32 elements along the row.  ISA 16-bit A layout: lane half 0 holds
// K {0..7,16..23}, half 1 holds K {8..15,24..31}.
static __device__ __forceinline__ v16bf load_a_at(const __bf16* __restrict__ ap,
                                                  int kt) {
  const __bf16* p = ap + (size_t)kt * 32;
  v8bf lo = *(const v8bf*)(p);
  v8bf hi = *(const v8bf*)(p + 16);
  return __builtin_shufflevector(lo, hi,
      0, 1, 2, 3, 4, 5, 6, 7, 8, 9, 10, 11, 12, 13, 14, 15);
}

// bp points at the packed B stream for one column tile (+ lane*16); k-tile
// stride is 512 bf16 (one 32x16 tile).
static __device__ __forceinline__ v16bf load_b_at(const __bf16* __restrict__ bp,
                                                  int kt) {
  return *(const v16bf*)(bp + (size_t)kt * 512);
}

static __device__ __forceinline__ v8f wmma_bf16(v16bf a, v16bf b, v8f c) {
  return __builtin_amdgcn_wmma_f32_16x16x32_bf16(
      /*neg_a=*/false, a, /*neg_b=*/false, b,
      /*c_mod=*/(short)0, c, /*reuse_a=*/false, /*reuse_b=*/false);
}

static __device__ __forceinline__ float sigmoidf_(float x) {
  return 1.0f / (1.0f + __expf(-x));
}

// ---- Prep kernels ----------------------------------------------------------

// Pack W (row-major [N, K1+K2], rows from W1|W2 concatenated along K) into
// the B-fragment layout, converting f32 -> bf16.  One block = one 32x16 tile.
__global__ void pack_b_kernel(__bf16* __restrict__ dst,
                              const float* __restrict__ W1, int K1,
                              const float* __restrict__ W2, int K2,
                              int KT) {
  const int kt = blockIdx.x % KT;
  const int nt = blockIdx.x / KT;
  const int lane = threadIdx.x;
  const int half = lane >> 4, r = lane & 15;
  const int n = nt * 16 + r;
  const int kb = kt * 32 + half * 8;
  __bf16* out = dst + ((size_t)nt * KT + kt) * 512 + lane * 16;
#pragma unroll
  for (int j = 0; j < 16; ++j) {
    const int k = kb + (j < 8 ? j : j + 8);
    const float v = (k < K1) ? W1[(size_t)n * K1 + k]
                             : W2[(size_t)n * K2 + (k - K1)];
    out[j] = (__bf16)v;
  }
}

__global__ void bias_kernel(float* __restrict__ b0, float* __restrict__ b1,
                            const float* __restrict__ bih0,
                            const float* __restrict__ bhh0,
                            const float* __restrict__ bih1,
                            const float* __restrict__ bhh1) {
  const int n = blockIdx.x * blockDim.x + threadIdx.x;
  if (n < 4 * HID) {
    b0[n] = bih0[n] + bhh0[n];
    b1[n] = bih1[n] + bhh1[n];
  }
}

// Initialize recurrent state from the h0/c0/init_output inputs.
__global__ void init_state_kernel(const float* __restrict__ h_in,
                                  const float* __restrict__ c_in,
                                  const float* __restrict__ out_in,
                                  float* __restrict__ h0f, float* __restrict__ h1f,
                                  float* __restrict__ c0f, float* __restrict__ c1f,
                                  __bf16* __restrict__ xcat0,
                                  __bf16* __restrict__ x1b0) {
  const int idx = blockIdx.x * blockDim.x + threadIdx.x;  // < BATCH*HID
  const int b = idx >> 10, k = idx & 1023;
  const float h0v = h_in[idx];
  const float h1v = h_in[BATCH * HID + idx];
  h0f[idx] = h0v;
  h1f[idx] = h1v;
  c0f[idx] = c_in[idx];
  c1f[idx] = c_in[BATCH * HID + idx];
  xcat0[(size_t)b * 3072 + 1024 + k] = (__bf16)out_in[idx];
  xcat0[(size_t)b * 3072 + 2048 + k] = (__bf16)h0v;
  x1b0[(size_t)b * 2048 + 1024 + k] = (__bf16)h1v;
}

// Embedding gather f32 -> bf16 into xcat[:, 0:1024].
__global__ void embed_kernel(const int* __restrict__ tok,
                             const float* __restrict__ W_emb,
                             __bf16* __restrict__ xR) {
  const int b = blockIdx.x;
  const int t = tok[b];
  const float* src = W_emb + (size_t)t * EMB;
  __bf16* dst = xR + (size_t)b * 3072;
  for (int k = threadIdx.x; k < EMB; k += blockDim.x)
    dst[k] = (__bf16)src[k];
}

// ---- Fused LSTM gate GEMM + cell update ------------------------------------
// grid = HID/16 blocks (one 16-wide h slice each), 256 threads = 8 waves.
// wave w: row-tile mt=w&3, gate-pair gh=w>>2 (gates 2gh, 2gh+1).  Each wave
// keeps one A stream + two gate B streams + two f32 accumulators, with TWO
// k-tiles of fragments in flight (depth-2 pipeline).  Gates are staged through
// LDS so the cell nonlinearity is fused; bf16 copies of h_new feed the next
// GEMMs.
__global__ __launch_bounds__(256) void lstm_gemm_kernel(
    const __bf16* __restrict__ X, int ldx, int K,
    const __bf16* __restrict__ Wp,
    const float* __restrict__ bias,
    float* __restrict__ Hst, float* __restrict__ Cst,
    __bf16* __restrict__ d1, int ld1,
    __bf16* __restrict__ d2, int ld2) {
  __shared__ float tileLds[16 * 256];  // 16 tiles of 16x16 f32
  const int tid = threadIdx.x, lane = tid & 31, w = tid >> 5;
  const int mt = w & 3, gh = w >> 2;
  const int g0 = 2 * gh, g1 = 2 * gh + 1;
  const int KT = K >> 5;
  const int half = lane >> 4, r = lane & 15;

  const __bf16* ap  = X + (size_t)(mt * 16 + r) * ldx + half * 8;
  const __bf16* bpA = Wp + (size_t)(g0 * (HID / 16) + blockIdx.x) * KT * 512 + lane * 16;
  const __bf16* bpB = Wp + (size_t)(g1 * (HID / 16) + blockIdx.x) * KT * 512 + lane * 16;

  v8f accA = {}, accB = {};
  // depth-2 pipeline: tiles kt and kt+1 resident, kt+2 loading
  v16bf a0  = load_a_at(ap, 0),  a1  = load_a_at(ap, 1);
  v16bf bA0 = load_b_at(bpA, 0), bA1 = load_b_at(bpA, 1);
  v16bf bB0 = load_b_at(bpB, 0), bB1 = load_b_at(bpB, 1);

  for (int kt = 0; kt + 2 < KT; ++kt) {
    if (kt + 8 < KT) {
      __builtin_prefetch(bpA + (size_t)(kt + 8) * 512, 0, 1);
      __builtin_prefetch(bpB + (size_t)(kt + 8) * 512, 0, 1);
    }
    v16bf a2  = load_a_at(ap, kt + 2);
    v16bf bA2 = load_b_at(bpA, kt + 2);
    v16bf bB2 = load_b_at(bpB, kt + 2);
    accA = wmma_bf16(a0, bA0, accA);
    accB = wmma_bf16(a0, bB0, accB);
    a0 = a1; a1 = a2;
    bA0 = bA1; bA1 = bA2;
    bB0 = bB1; bB1 = bB2;
  }
  accA = wmma_bf16(a0, bA0, accA);
  accB = wmma_bf16(a0, bB0, accB);
  accA = wmma_bf16(a1, bA1, accA);
  accB = wmma_bf16(a1, bB1, accB);

  const float bvA = bias[(g0 * (HID / 16) + blockIdx.x) * 16 + r];
  const float bvB = bias[(g1 * (HID / 16) + blockIdx.x) * 16 + r];
#pragma unroll
  for (int i = 0; i < 8; ++i) {
    tileLds[(g0 * 4 + mt) * 256 + (i + 8 * half) * 16 + r] = accA[i] + bvA;
    tileLds[(g1 * 4 + mt) * 256 + (i + 8 * half) * 16 + r] = accB[i] + bvB;
  }
  __syncthreads();

  const int hcol0 = blockIdx.x * 16;
  for (int p = tid; p < BATCH * 16; p += 256) {
    const int bb = p >> 4, hl = p & 15;
    const int mtt = bb >> 4, rl = bb & 15;
    const float gi = tileLds[(0 * 4 + mtt) * 256 + rl * 16 + hl];
    const float gf = tileLds[(1 * 4 + mtt) * 256 + rl * 16 + hl];
    const float gg = tileLds[(2 * 4 + mtt) * 256 + rl * 16 + hl];
    const float go = tileLds[(3 * 4 + mtt) * 256 + rl * 16 + hl];
    const int col = hcol0 + hl;
    const float c_old = Cst[(size_t)bb * HID + col];
    const float cn = sigmoidf_(gf) * c_old + sigmoidf_(gi) * tanhf(gg);
    const float hn = sigmoidf_(go) * tanhf(cn);
    Cst[(size_t)bb * HID + col] = cn;
    Hst[(size_t)bb * HID + col] = hn;
    const __bf16 hb = (__bf16)hn;
    d1[(size_t)bb * ld1 + col] = hb;
    d2[(size_t)bb * ld2 + col] = hb;
  }
}

// ---- Plain WMMA GEMM, f32 output (attention linear_in) ---------------------
// grid = N/16, 128 threads = 4 waves (row tiles), depth-2 pipelined.
__global__ __launch_bounds__(128) void gemm_f32out_kernel(
    const __bf16* __restrict__ X, int ldx, int K,
    const __bf16* __restrict__ Wp,
    float* __restrict__ C, int ldc) {
  const int lane = threadIdx.x & 31, mt = threadIdx.x >> 5;
  const int KT = K >> 5, nt = blockIdx.x;
  const int half = lane >> 4, r = lane & 15;
  const __bf16* ap = X + (size_t)(mt * 16 + r) * ldx + half * 8;
  const __bf16* bp = Wp + (size_t)nt * KT * 512 + lane * 16;

  v8f acc = {};
  v16bf a0 = load_a_at(ap, 0), a1 = load_a_at(ap, 1);
  v16bf b0 = load_b_at(bp, 0), b1 = load_b_at(bp, 1);
  for (int kt = 0; kt + 2 < KT; ++kt) {
    if (kt + 8 < KT) __builtin_prefetch(bp + (size_t)(kt + 8) * 512, 0, 1);
    v16bf a2 = load_a_at(ap, kt + 2);
    v16bf b2 = load_b_at(bp, kt + 2);
    acc = wmma_bf16(a0, b0, acc);
    a0 = a1; a1 = a2;
    b0 = b1; b1 = b2;
  }
  acc = wmma_bf16(a0, b0, acc);
  acc = wmma_bf16(a1, b1, acc);

#pragma unroll
  for (int i = 0; i < 8; ++i) {
    const int row = mt * 16 + i + 8 * half;
    C[(size_t)row * ldc + nt * 16 + r] = acc[i];
  }
}

// ---- WMMA GEMM + tanh, dual f32/bf16 output (attention linear_out) ---------
__global__ __launch_bounds__(128) void gemm_tanh_kernel(
    const __bf16* __restrict__ X, int ldx, int K,
    const __bf16* __restrict__ Wp,
    float* __restrict__ Of,           // [B, HID] slice of d_out outputs
    __bf16* __restrict__ Ob, int ldob) {
  const int lane = threadIdx.x & 31, mt = threadIdx.x >> 5;
  const int KT = K >> 5, nt = blockIdx.x;
  const int half = lane >> 4, r = lane & 15;
  const __bf16* ap = X + (size_t)(mt * 16 + r) * ldx + half * 8;
  const __bf16* bp = Wp + (size_t)nt * KT * 512 + lane * 16;

  v8f acc = {};
  v16bf a0 = load_a_at(ap, 0), a1 = load_a_at(ap, 1);
  v16bf b0 = load_b_at(bp, 0), b1 = load_b_at(bp, 1);
  for (int kt = 0; kt + 2 < KT; ++kt) {
    if (kt + 8 < KT) __builtin_prefetch(bp + (size_t)(kt + 8) * 512, 0, 1);
    v16bf a2 = load_a_at(ap, kt + 2);
    v16bf b2 = load_b_at(bp, kt + 2);
    acc = wmma_bf16(a0, b0, acc);
    a0 = a1; a1 = a2;
    b0 = b1; b1 = b2;
  }
  acc = wmma_bf16(a0, b0, acc);
  acc = wmma_bf16(a1, b1, acc);

#pragma unroll
  for (int i = 0; i < 8; ++i) {
    const int row = mt * 16 + i + 8 * half;
    const int col = nt * 16 + r;
    const float v = tanhf(acc[i]);
    Of[(size_t)row * HID + col] = v;
    Ob[(size_t)row * ldob + col] = (__bf16)v;
  }
}

// ---- Attention: scores -> softmax -> weighted context ----------------------
// One block per batch row, 256 threads = 8 waves.
__global__ __launch_bounds__(256) void attn_kernel(
    const float* __restrict__ ctx,     // [B, S, H]
    const float* __restrict__ tgt,     // [B, H]
    __bf16* __restrict__ wctx_dst,     // cat2[:, 0:H], ld = 2*H
    float* __restrict__ attn_out) {    // [B, S] slice of d_out attns
  __shared__ float sc[SRC];
  __shared__ float av[SRC];
  const int b = blockIdx.x;
  const int tid = threadIdx.x, lane = tid & 31, w = tid >> 5;
  const float* cb = ctx + (size_t)b * SRC * HID;
  const float* tb = tgt + (size_t)b * HID;

  for (int s = w * 6; s < w * 6 + 6; ++s) {
    float acc = 0.f;
    for (int h = lane; h < HID; h += 32)
      acc += cb[(size_t)s * HID + h] * tb[h];
#pragma unroll
    for (int m = 16; m >= 1; m >>= 1) acc += __shfl_xor(acc, m, 32);
    if (lane == 0) sc[s] = acc;
  }
  __syncthreads();

  if (tid < SRC) {
    float mx = -1e30f;
    for (int i = 0; i < SRC; ++i) mx = fmaxf(mx, sc[i]);
    float sum = 0.f;
    for (int i = 0; i < SRC; ++i) sum += __expf(sc[i] - mx);
    const float a = __expf(sc[tid] - mx) / sum;
    av[tid] = a;
    attn_out[b * SRC + tid] = a;
  }
  __syncthreads();

  for (int h = tid; h < HID; h += blockDim.x) {
    float acc = 0.f;
#pragma unroll 4
    for (int s = 0; s < SRC; ++s) acc += av[s] * cb[(size_t)s * HID + h];
    wctx_dst[(size_t)b * (2 * HID) + h] = (__bf16)acc;
  }
}

// ---------------------------------------------------------------------------

extern "C" void kernel_launch(void* const* d_in, const int* in_sizes, int n_in,
                              void* d_out, int out_size, void* d_ws, size_t ws_size,
                              hipStream_t stream) {
  const int*   tokens   = (const int*)d_in[0];
  const float* h_in     = (const float*)d_in[1];
  const float* c_in     = (const float*)d_in[2];
  const float* context  = (const float*)d_in[3];
  const float* init_out = (const float*)d_in[4];
  const float* W_emb    = (const float*)d_in[5];
  const float* Wih0     = (const float*)d_in[6];
  const float* Whh0     = (const float*)d_in[7];
  const float* bih0     = (const float*)d_in[8];
  const float* bhh0     = (const float*)d_in[9];
  const float* Wih1     = (const float*)d_in[10];
  const float* Whh1     = (const float*)d_in[11];
  const float* bih1     = (const float*)d_in[12];
  const float* bhh1     = (const float*)d_in[13];
  const float* Wa_in    = (const float*)d_in[14];
  const float* Wa_out   = (const float*)d_in[15];

  // ---- workspace carve ----
  char* ws = (char*)d_ws;
  auto carve = [&](size_t bytes) -> char* {
    char* p = ws;
    ws += (bytes + 255) & ~(size_t)255;
    return p;
  };
  __bf16* W0p    = (__bf16*)carve((size_t)3072 * 4096 * 2);
  __bf16* W1p    = (__bf16*)carve((size_t)2048 * 4096 * 2);
  __bf16* WaInP  = (__bf16*)carve((size_t)1024 * 1024 * 2);
  __bf16* WaOutP = (__bf16*)carve((size_t)2048 * 1024 * 2);
  float*  b0     = (float*)carve(4 * HID * 4);
  float*  b1     = (float*)carve(4 * HID * 4);
  __bf16* xcat[2] = {(__bf16*)carve((size_t)BATCH * 3072 * 2),
                     (__bf16*)carve((size_t)BATCH * 3072 * 2)};
  __bf16* x1b[2]  = {(__bf16*)carve((size_t)BATCH * 2048 * 2),
                     (__bf16*)carve((size_t)BATCH * 2048 * 2)};
  __bf16* cat2    = (__bf16*)carve((size_t)BATCH * 2048 * 2);
  float*  h0f     = (float*)carve((size_t)BATCH * HID * 4);
  float*  h1f     = (float*)carve((size_t)BATCH * HID * 4);
  float*  c0f     = (float*)carve((size_t)BATCH * HID * 4);
  float*  c1f     = (float*)carve((size_t)BATCH * HID * 4);
  float*  tgt     = (float*)carve((size_t)BATCH * HID * 4);

  // ---- output carve: outputs[T,B,H] | hT[2,B,H] | cT[2,B,H] | attns[T,B,S]
  float* out_f = (float*)d_out;
  float* hT    = out_f + (size_t)T_STEPS * BATCH * HID;
  float* cT    = hT + 2 * BATCH * HID;
  float* attns = cT + 2 * BATCH * HID;

  // ---- weight prep (every call; deterministic) ----
  pack_b_kernel<<<256 * 96, 32, 0, stream>>>(W0p, Wih0, 2048, Whh0, 1024, 96);
  pack_b_kernel<<<256 * 64, 32, 0, stream>>>(W1p, Wih1, 1024, Whh1, 1024, 64);
  pack_b_kernel<<<64 * 32, 32, 0, stream>>>(WaInP, Wa_in, 1024, nullptr, 0, 32);
  pack_b_kernel<<<64 * 64, 32, 0, stream>>>(WaOutP, Wa_out, 2048, nullptr, 0, 64);
  bias_kernel<<<(4 * HID + 255) / 256, 256, 0, stream>>>(b0, b1, bih0, bhh0, bih1, bhh1);
  init_state_kernel<<<(BATCH * HID) / 256, 256, 0, stream>>>(
      h_in, c_in, init_out, h0f, h1f, c0f, c1f, xcat[0], x1b[0]);

  // ---- time loop (sequential dependency; single stream serializes) ----
  for (int t = 0; t < T_STEPS; ++t) {
    const int p = t & 1;
    __bf16* xR = xcat[p];       // read:  [emb_t | out_{t-1} | h0_{t-1}]
    __bf16* xW = xcat[1 - p];   // write: h0_t (cols 2048..), out_t (cols 1024..)
    __bf16* x1R = x1b[p];       // read:  [h0_t | h1_{t-1}]
    __bf16* x1W = x1b[1 - p];   // write: h1_t (cols 1024..)

    embed_kernel<<<BATCH, 256, 0, stream>>>(tokens + (size_t)t * BATCH, W_emb, xR);

    // layer 0: gates = [emb,out,h0] @ W0cat^T; fused cell update
    lstm_gemm_kernel<<<HID / 16, 256, 0, stream>>>(
        xR, 3072, 3072, W0p, b0, h0f, c0f,
        /*d1=*/x1R, /*ld1=*/2048,            // h0_t -> layer-1 input (this step)
        /*d2=*/xW + 2048, /*ld2=*/3072);     // h0_t -> next step's xcat

    // layer 1: gates = [h0_t, h1_{t-1}] @ W1cat^T
    lstm_gemm_kernel<<<HID / 16, 256, 0, stream>>>(
        x1R, 2048, 2048, W1p, b1, h1f, c1f,
        /*d1=*/cat2 + 1024, /*ld1=*/2048,    // rnn_out -> attn concat (+ attnIn A)
        /*d2=*/x1W + 1024, /*ld2=*/2048);    // h1_t -> next step's x1cat

    // tgt = rnn_out @ Wa_in^T
    gemm_f32out_kernel<<<HID / 16, 128, 0, stream>>>(
        cat2 + 1024, 2048, 1024, WaInP, tgt, HID);

    // softmax(context . tgt) -> wctx; attn weights to d_out
    attn_kernel<<<BATCH, 256, 0, stream>>>(
        context, tgt, cat2, attns + (size_t)t * BATCH * SRC);

    // out_t = tanh([wctx, rnn_out] @ Wa_out^T)
    gemm_tanh_kernel<<<HID / 16, 128, 0, stream>>>(
        cat2, 2048, 2048, WaOutP,
        out_f + (size_t)t * BATCH * HID,     // outputs[t]
        xW + 1024, 3072);                    // out_t -> next step's xcat
  }

  // ---- final states ----
  const size_t bh = (size_t)BATCH * HID * sizeof(float);
  hipMemcpyAsync(hT, h0f, bh, hipMemcpyDeviceToDevice, stream);
  hipMemcpyAsync(hT + BATCH * HID, h1f, bh, hipMemcpyDeviceToDevice, stream);
  hipMemcpyAsync(cT, c0f, bh, hipMemcpyDeviceToDevice, stream);
  hipMemcpyAsync(cT + BATCH * HID, c1f, bh, hipMemcpyDeviceToDevice, stream);
}
